// MLPModelPredictor_59691455480298
// MI455X (gfx1250) — matrix operand order repaired and checked by reading, not compiled, not run on hardware
//
#include <hip/hip_runtime.h>
#include <hip/hip_bf16.h>

typedef __attribute__((ext_vector_type(16))) __bf16 v16bf;
typedef __attribute__((ext_vector_type(8)))  __bf16 v8bf;
typedef __attribute__((ext_vector_type(8)))  float  v8f;

#define T_DIM 1024
#define B_DIM 256
#define F_DIM 100
#define H_DIM 512
#define R_DIM 32
#define O_DIM 10

#define WAVES 4
#define TPB   (WAVES * 32)
#define TROWS (WAVES * 16)          // 64 t-rows per block
#define TILES_PER_B (T_DIM / TROWS) // 16

// ---- LDS layout (byte offsets, all 32B aligned) ----
#define OFF_BUFA 0                  // 512*32 bf16 = 32 KB : sw0_t / sw1_t   [h][k]
#define OFF_BUFB 32768              // 32*512 bf16 = 32 KB : w1_t/w2_t/w3_t  [n][k]
#define OFF_BIAS 65536              // 512 f32 = 2 KB
#define OFF_MINI 67584              // 4 waves * 16*32 bf16 = 4 KB
#define OFF_HST  71680              // 4 waves * 16*528 bf16 = 66 KB
#define HST_STRIDE 528
#define LDS_BYTES (OFF_HST + WAVES * 16 * HST_STRIDE * 2)  // 139264

#define WMMA_BF16(A, Bv, C) \
  __builtin_amdgcn_wmma_f32_16x16x32_bf16(false, (A), false, (Bv), (short)0, (C), false, false)

union AFrag { v16bf v; v8bf p[2]; __bf16 e[16]; };

__global__ __launch_bounds__(TPB)
void mlp_fused_wmma(const float* __restrict__ x,  const float* __restrict__ w1,
                    const float* __restrict__ b1, const float* __restrict__ sw0,
                    const float* __restrict__ w2, const float* __restrict__ b2,
                    const float* __restrict__ sw1,const float* __restrict__ w3,
                    const float* __restrict__ b3, float* __restrict__ out) {
  extern __shared__ __align__(32) char smem[];
  __bf16* bufA = (__bf16*)(smem + OFF_BUFA);   // shared-weight (transposed) [N][K]
  __bf16* bufB = (__bf16*)(smem + OFF_BUFB);   // per-b weight (transposed) [N][K]
  float*  bias = (float*) (smem + OFF_BIAS);

  const int tid  = threadIdx.x;
  const int wave = tid >> 5;
  const int lane = tid & 31;
  const int half = lane >> 4;   // 0: lanes 0-15, 1: lanes 16-31
  const int col  = lane & 15;   // A-row / B-column / C-column index

  __bf16* mini = (__bf16*)(smem + OFF_MINI) + wave * (16 * 32);
  __bf16* hst  = (__bf16*)(smem + OFF_HST)  + wave * (16 * HST_STRIDE);

  const int b  = blockIdx.x >> 4;                       // batch index
  const int t0 = (blockIdx.x & 15) * TROWS + wave * 16; // this wave's first t-row

  // ===================== Phase A: load w1_t + sw0_t + b1 =====================
  for (int i = tid; i < 32 * 128; i += TPB) bufB[i] = (__bf16)0.f;  // zero-pad F->128
  __syncthreads();
  {
    const float* w1b = w1 + (size_t)b * F_DIM * R_DIM;           // [F][R]
    for (int i = tid; i < F_DIM * R_DIM; i += TPB) {
      int f = i >> 5, r = i & 31;
      bufB[r * 128 + f] = (__bf16)w1b[i];                        // w1_t[r][f]
    }
    for (int i = tid; i < R_DIM * H_DIM; i += TPB) {             // sw0 [R][H]
      int k = i >> 9, h = i & 511;
      bufA[h * 32 + k] = (__bf16)sw0[i];                         // sw0_t[h][k]
    }
    const float* b1b = b1 + (size_t)b * H_DIM;
    for (int i = tid; i < H_DIM; i += TPB) bias[i] = b1b[i];
  }
  __syncthreads();

  // ===================== Stage 1: h1[16x32] = x @ w1[b] =====================
  v8f acc0 = {}, acc1 = {};
  {
    const float* xr = x + (((size_t)(t0 + col)) * B_DIM + b) * F_DIM; // A row = col
    for (int k0 = 0; k0 < 128; k0 += 32) {
      AFrag a;
      const int base = k0 + half * 16;
#pragma unroll
      for (int j = 0; j < 8; ++j) {
        int f0 = base + j, f1 = base + 8 + j;
        float v0 = (f0 < F_DIM) ? xr[f0] : 0.f;
        float v1 = (f1 < F_DIM) ? xr[f1] : 0.f;
        v0 = (v0 == v0) ? v0 : 0.f;  // nan_to_num
        v1 = (v1 == v1) ? v1 : 0.f;
        a.e[j]     = (__bf16)v0;
        a.e[8 + j] = (__bf16)v1;
      }
      v16bf bf0 = *(const v16bf*)(bufB + (0 * 16 + col) * 128 + k0 + half * 16);
      v16bf bf1 = *(const v16bf*)(bufB + (1 * 16 + col) * 128 + k0 + half * 16);
      acc0 = WMMA_BF16(a.v, bf0, acc0);
      acc1 = WMMA_BF16(a.v, bf1, acc1);
    }
  }
#pragma unroll
  for (int r = 0; r < 8; ++r) {    // C layout -> row-major bf16 mini[16][32]
    int m = r + 8 * half;
    mini[m * 32 +      col] = (__bf16)acc0[r];
    mini[m * 32 + 16 + col] = (__bf16)acc1[r];
  }

  // ============ Stage 2: h = relu(h1 @ sw0 + b1) -> hst[16][512] ============
  {
    AFrag a;   // K=32 fragment, reused across all 32 n-tiles
    const __bf16* ap = mini + col * 32;
    a.p[0] = *(const v8bf*)(ap + half * 16);
    a.p[1] = *(const v8bf*)(ap + half * 16 + 8);
    for (int n = 0; n < 32; ++n) {
      float bv = bias[n * 16 + col];
      v8f c = {bv, bv, bv, bv, bv, bv, bv, bv};
      v16bf bfrag = *(const v16bf*)(bufA + (n * 16 + col) * 32 + half * 16);
      c = WMMA_BF16(a.v, bfrag, c);
#pragma unroll
      for (int r = 0; r < 8; ++r) {
        float v = c[r]; v = v > 0.f ? v : 0.f;   // relu
        hst[(r + 8 * half) * HST_STRIDE + n * 16 + col] = (__bf16)v;
      }
    }
  }
  __syncthreads();

  // ===================== Phase B: load w2_t + sw1_t + b2 =====================
  {
    const float* w2b = w2 + (size_t)b * H_DIM * R_DIM;           // [H][R]
    for (int i = tid; i < H_DIM * R_DIM; i += TPB) {
      int h = i >> 5, r = i & 31;
      bufB[r * 512 + h] = (__bf16)w2b[i];                        // w2_t[r][h]
    }
    for (int i = tid; i < R_DIM * H_DIM; i += TPB) {             // sw1 [R][H]
      int k = i >> 9, h = i & 511;
      bufA[h * 32 + k] = (__bf16)sw1[i];                         // sw1_t[h][k]
    }
    const float* b2b = b2 + (size_t)b * H_DIM;
    for (int i = tid; i < H_DIM; i += TPB) bias[i] = b2b[i];
  }
  __syncthreads();

  // ===================== Stage 3: h2[16x32] = h @ w2[b] =====================
  {
    v8f a0 = {}, a1 = {};
    for (int k0 = 0; k0 < H_DIM; k0 += 32) {
      AFrag a;
      const __bf16* ap = hst + col * HST_STRIDE + k0;
      a.p[0] = *(const v8bf*)(ap + half * 16);
      a.p[1] = *(const v8bf*)(ap + half * 16 + 8);
      v16bf bf0 = *(const v16bf*)(bufB + (0 * 16 + col) * 512 + k0 + half * 16);
      v16bf bf1 = *(const v16bf*)(bufB + (1 * 16 + col) * 512 + k0 + half * 16);
      a0 = WMMA_BF16(a.v, bf0, a0);
      a1 = WMMA_BF16(a.v, bf1, a1);
    }
#pragma unroll
    for (int r = 0; r < 8; ++r) {
      int m = r + 8 * half;
      mini[m * 32 +      col] = (__bf16)a0[r];
      mini[m * 32 + 16 + col] = (__bf16)a1[r];
    }
  }

  // ============ Stage 4: h = relu(h2 @ sw1 + b2) -> hst[16][512] ============
  {
    AFrag a;
    const __bf16* ap = mini + col * 32;
    a.p[0] = *(const v8bf*)(ap + half * 16);
    a.p[1] = *(const v8bf*)(ap + half * 16 + 8);
    for (int n = 0; n < 32; ++n) {
      float bv = bias[n * 16 + col];
      v8f c = {bv, bv, bv, bv, bv, bv, bv, bv};
      v16bf bfrag = *(const v16bf*)(bufA + (n * 16 + col) * 32 + half * 16);
      c = WMMA_BF16(a.v, bfrag, c);
#pragma unroll
      for (int r = 0; r < 8; ++r) {
        float v = c[r]; v = v > 0.f ? v : 0.f;
        hst[(r + 8 * half) * HST_STRIDE + n * 16 + col] = (__bf16)v;
      }
    }
  }
  __syncthreads();

  // ===================== Phase C: load w3_t (O padded to 16) + b3 ===========
  for (int i = tid; i < 16 * H_DIM; i += TPB) bufB[i] = (__bf16)0.f;
  __syncthreads();
  {
    const float* w3b = w3 + (size_t)b * H_DIM * O_DIM;           // [H][O]
    for (int i = tid; i < H_DIM * O_DIM; i += TPB) {
      int h = i / O_DIM, o = i - h * O_DIM;
      bufB[o * 512 + h] = (__bf16)w3b[i];                        // w3_t[o][h]
    }
    const float* b3b = b3 + (size_t)b * O_DIM;
    for (int i = tid; i < 16; i += TPB) bias[i] = (i < O_DIM) ? b3b[i] : 0.f;
  }
  __syncthreads();

  // ===================== Stage 5: out = h @ w3[b] + b3 ======================
  {
    float bv = bias[col];
    v8f c = {bv, bv, bv, bv, bv, bv, bv, bv};
    for (int k0 = 0; k0 < H_DIM; k0 += 32) {
      AFrag a;
      const __bf16* ap = hst + col * HST_STRIDE + k0;
      a.p[0] = *(const v8bf*)(ap + half * 16);
      a.p[1] = *(const v8bf*)(ap + half * 16 + 8);
      v16bf bfrag = *(const v16bf*)(bufB + col * 512 + k0 + half * 16);
      c = WMMA_BF16(a.v, bfrag, c);
    }
    if (col < O_DIM) {
#pragma unroll
      for (int r = 0; r < 8; ++r) {
        int t = t0 + r + 8 * half;
        out[((size_t)t * B_DIM + b) * O_DIM + col] = c[r];
      }
    }
  }
}

extern "C" void kernel_launch(void* const* d_in, const int* in_sizes, int n_in,
                              void* d_out, int out_size, void* d_ws, size_t ws_size,
                              hipStream_t stream) {
  const float* x   = (const float*)d_in[0];
  const float* w1  = (const float*)d_in[1];
  const float* b1  = (const float*)d_in[2];
  const float* sw0 = (const float*)d_in[3];
  const float* w2  = (const float*)d_in[4];
  const float* b2  = (const float*)d_in[5];
  const float* sw1 = (const float*)d_in[6];
  const float* w3  = (const float*)d_in[7];
  const float* b3  = (const float*)d_in[8];
  float* out = (float*)d_out;

  (void)hipFuncSetAttribute((const void*)mlp_fused_wmma,
                            hipFuncAttributeMaxDynamicSharedMemorySize, LDS_BYTES);

  dim3 grid(B_DIM * TILES_PER_B);   // 4096 blocks: (b, t-tile)
  mlp_fused_wmma<<<grid, TPB, LDS_BYTES, stream>>>(x, w1, b1, sw0, w2, b2, sw1, w3, b3, out);
}
